// MiniGPT_78469052498074
// MI455X (gfx1250) — matrix operand order repaired and checked by reading, not compiled, not run on hardware
//
#include <hip/hip_runtime.h>
#include <hip/hip_bf16.h>
#include <math.h>

typedef __attribute__((ext_vector_type(16))) _Float16 v16h;
typedef __attribute__((ext_vector_type(8)))  _Float16 v8h;
typedef __attribute__((ext_vector_type(4)))  _Float16 v4h;
typedef __attribute__((ext_vector_type(8)))  float    v8f;
typedef __attribute__((ext_vector_type(4)))  unsigned v4u;
typedef __attribute__((ext_vector_type(8)))  int      v8i;
typedef __attribute__((ext_vector_type(4)))  int      v4i;

#if defined(__has_builtin)
#if __has_builtin(__builtin_amdgcn_tensor_load_to_lds) && __has_builtin(__builtin_amdgcn_s_wait_tensorcnt)
#define USE_TDM 1
#endif
#endif
#ifndef USE_TDM
#define USE_TDM 0
#endif

#if USE_TDM
// ---------------------------------------------------------------------------
// Issue one TDM 2-D tile load (f16, 32 cols x 128 rows) into LDS.
//   D# per CDNA5 ISA:
//   group0: count=1 | lds_addr | global_addr | type=2
//   group1: data_size=2B, pad_enable, pad_interval=16DW(64B), pad_amount=4DW
//           (LDS row stride = 80B = 40 f16, matching the fragment readers),
//           tensor_dim0=32, tensor_dim1=rows_remaining (OOB rows -> zeros),
//           tile 32x128, dim0 stride = ld.
// ---------------------------------------------------------------------------
__device__ __forceinline__ void tdm_tile_load(unsigned lds_addr,
    const _Float16* base, int ld, int kk, int row0, int rows_total)
{
    unsigned long long ga = (unsigned long long)(uintptr_t)base
        + ((unsigned long long)(unsigned)row0 * (unsigned)ld + (unsigned)kk) * 2ull;
    int rows = rows_total - row0; if (rows < 0) rows = 0;
    v4u g0 = { 1u, lds_addr, (unsigned)ga,
               (unsigned)((ga >> 32) & 0x1FFFFFFu) | (2u << 30) };
    v8i g1 = { (int)((1u << 16) | (1u << 20) | (3u << 22) | (3u << 25)),
               (int)(32u << 16),                         // tensor_dim0 = 32
               (int)(((unsigned)rows & 0xFFFFu) << 16),  // tensor_dim1 lo
               (int)(((unsigned)rows >> 16) | (32u << 16)), // dim1 hi | tile_dim0
               (int)128,                                 // tile_dim1 = 128
               (int)(unsigned)ld,                        // dim0 stride lo
               0, 0 };
    v4i g2 = {0, 0, 0, 0}, g3 = {0, 0, 0, 0};
#if __clang_major__ >= 23
    v8i gz = {0, 0, 0, 0, 0, 0, 0, 0};
    __builtin_amdgcn_tensor_load_to_lds(g0, g1, g2, g3, gz, 0);
#else
    __builtin_amdgcn_tensor_load_to_lds(g0, g1, g2, g3, 0);
#endif
}
#endif

// ---------------------------------------------------------------------------
// Batched WMMA GEMM:  C[M,N] = epilogue( A[M,K] * B[N,K]^T ), A/B f16 global.
//   Block tile 128x128, 8 waves (4Mx2N), wave tile 32x64 (2x4 frags), K-step
//   32, double-buffered LDS. Tile staging via TDM (tensor_load_to_lds, wave 0,
//   overlapped with WMMA, fenced by s_wait_tensorcnt + barrier) or cooperative
//   v8h copies when the TDM builtin is unavailable.
//   op bits: 1=+bias[n], 2=exact GELU, 4=+residual R(f32, layout of C)
//   cfmt: 0 -> C f32, 1 -> C f16.  ctrans: store C[n*ldc+m] (V^T projection).
//   Batch z = (bz=z/nh, hz=z%nh) with independent strides per matrix.
// ---------------------------------------------------------------------------
__global__ __launch_bounds__(256) void gemm_h(
    const _Float16* __restrict__ A, const _Float16* __restrict__ Bm,
    const float* __restrict__ bias, const float* __restrict__ R,
    void* __restrict__ Cv,
    int M, int N, int K, int lda, int ldb, int ldc,
    long long sa0, long long sa1, long long sb0, long long sb1,
    long long sc0, long long sc1,
    int nh, int op, int cfmt, int ctrans)
{
    __shared__ _Float16 As[2][128 * 40];
    __shared__ _Float16 Bs[2][128 * 40];

    const int z  = blockIdx.z;
    const int bz = z / nh, hz = z % nh;
    A  += (size_t)bz * sa0 + (size_t)hz * sa1;
    Bm += (size_t)bz * sb0 + (size_t)hz * sb1;
    float*    Cf = (float*)Cv    + (size_t)bz * sc0 + (size_t)hz * sc1;
    _Float16* Ch = (_Float16*)Cv + (size_t)bz * sc0 + (size_t)hz * sc1;
    if (R) R += (size_t)bz * sc0 + (size_t)hz * sc1;

    const int tid  = threadIdx.x;
    const int lane = tid & 31;
    const int wid  = tid >> 5;
    const int wm   = wid & 3;
    const int wn   = wid >> 2;
    const int ln16 = lane & 15;
    const int half = lane >> 4;

    const int m0 = blockIdx.y * 128;
    const int n0 = blockIdx.x * 128;

    const v8f vzero = {0.f, 0.f, 0.f, 0.f, 0.f, 0.f, 0.f, 0.f};
    v8f acc[2][4];
    #pragma unroll
    for (int i = 0; i < 2; ++i)
        #pragma unroll
        for (int j = 0; j < 4; ++j) acc[i][j] = vzero;

#if USE_TDM
    const unsigned ldsA[2] = { (unsigned)(uintptr_t)&As[0][0],
                               (unsigned)(uintptr_t)&As[1][0] };
    const unsigned ldsB[2] = { (unsigned)(uintptr_t)&Bs[0][0],
                               (unsigned)(uintptr_t)&Bs[1][0] };
    if (wid == 0) {
        tdm_tile_load(ldsA[0], A,  lda, 0, m0, M);
        tdm_tile_load(ldsB[0], Bm, ldb, 0, n0, N);
        __builtin_amdgcn_s_wait_tensorcnt(0);
    }
    __syncthreads();
#else
    const int srow = tid >> 2;            // 64 rows per chunk
    const int scol = (tid & 3) * 8;       // 8 f16 (16 B) per access
    auto load_tile = [&](int kk, v8h* ar, v8h* br) {
        #pragma unroll
        for (int c = 0; c < 2; ++c) {
            int row = srow + c * 64;
            int gm = m0 + row; gm = (gm < M) ? gm : (M - 1);
            int gn = n0 + row; gn = (gn < N) ? gn : (N - 1);
            ar[c] = *(const v8h*)(A  + (size_t)gm * lda + kk + scol);
            br[c] = *(const v8h*)(Bm + (size_t)gn * ldb + kk + scol);
        }
    };
    auto store_tile = [&](int buf, const v8h* ar, const v8h* br) {
        #pragma unroll
        for (int c = 0; c < 2; ++c) {
            int row = srow + c * 64;
            *(v8h*)(&As[buf][row * 40 + scol]) = ar[c];
            *(v8h*)(&Bs[buf][row * 40 + scol]) = br[c];
        }
    };
    v8h ar[2], br[2];
    load_tile(0, ar, br);
    store_tile(0, ar, br);
    __syncthreads();
#endif

    int cur = 0;
    for (int kk = 0; kk < K; kk += 32) {
        const bool more = (kk + 32) < K;
#if USE_TDM
        if (more && wid == 0) {
            tdm_tile_load(ldsA[cur ^ 1], A,  lda, kk + 32, m0, M);
            tdm_tile_load(ldsB[cur ^ 1], Bm, ldb, kk + 32, n0, N);
        }
#else
        v8h ar2[2], br2[2];
        if (more) {
            load_tile(kk + 32, ar2, br2);
            if (kk + 64 < K) {
                __builtin_prefetch(A  + (size_t)(m0 + srow) * lda + kk + 64, 0, 0);
                __builtin_prefetch(Bm + (size_t)(n0 + srow) * ldb + kk + 64, 0, 0);
            }
        }
#endif
        // ---- fragments per ISA 7.12.2 VGPR layouts ----
        v16h af[2], bfr[4];
        #pragma unroll
        for (int i = 0; i < 2; ++i) {
            const _Float16* ap = &As[cur][(wm * 32 + i * 16 + ln16) * 40];
            v8h alo = *(const v8h*)(ap + half * 8);
            v8h ahi = *(const v8h*)(ap + 16 + half * 8);
            af[i] = __builtin_shufflevector(alo, ahi,
                    0,1,2,3,4,5,6,7,8,9,10,11,12,13,14,15);
        }
        #pragma unroll
        for (int j = 0; j < 4; ++j) {
            const _Float16* bp = &Bs[cur][(wn * 64 + j * 16 + ln16) * 40];
            v8h blo = *(const v8h*)(bp + half * 8);
            v8h bhi = *(const v8h*)(bp + 16 + half * 8);
            bfr[j] = __builtin_shufflevector(blo, bhi,
                    0,1,2,3,4,5,6,7,8,9,10,11,12,13,14,15);
        }
        #pragma unroll
        for (int i = 0; i < 2; ++i)
            #pragma unroll
            for (int j = 0; j < 4; ++j)
                acc[i][j] = __builtin_amdgcn_wmma_f32_16x16x32_f16(
                    false, af[i], false, bfr[j], (short)0, acc[i][j], false, false);

        if (more) {
#if USE_TDM
            if (wid == 0) __builtin_amdgcn_s_wait_tensorcnt(0);
#else
            store_tile(cur ^ 1, ar2, br2);
#endif
            __syncthreads();
            cur ^= 1;
        }
    }

    // ---- epilogue: lane ln16 = n, vgpr r: m = half*8 + r ----
    #pragma unroll
    for (int i = 0; i < 2; ++i) {
        #pragma unroll
        for (int j = 0; j < 4; ++j) {
            #pragma unroll
            for (int r = 0; r < 8; ++r) {
                int m = m0 + wm * 32 + i * 16 + half * 8 + r;
                int n = n0 + wn * 64 + j * 16 + ln16;
                if (m < M && n < N) {
                    float v = acc[i][j][r];
                    if (op & 1) v += bias[n];
                    if (op & 2) v = 0.5f * v * (1.0f + erff(v * 0.70710678118654752f));
                    if (op & 4) v += R[(size_t)m * ldc + n];
                    size_t idx = ctrans ? ((size_t)n * ldc + m) : ((size_t)m * ldc + n);
                    if (cfmt) Ch[idx] = (_Float16)v;
                    else      Cf[idx] = v;
                }
            }
        }
    }
}

// ---------------------------------------------------------------------------
// f32 -> f16 bulk conversion (vectorized, grid-stride), n4 = elems/4
// ---------------------------------------------------------------------------
__global__ __launch_bounds__(256) void cvt_k(const float* __restrict__ s,
    _Float16* __restrict__ d, long long n4)
{
    long long i = (long long)blockIdx.x * 256 + threadIdx.x;
    const long long stride = (long long)gridDim.x * 256;
    for (; i < n4; i += stride) {
        float4 f = ((const float4*)s)[i];
        v4h h = { (_Float16)f.x, (_Float16)f.y, (_Float16)f.z, (_Float16)f.w };
        ((v4h*)d)[i] = h;
    }
}

// ---------------------------------------------------------------------------
// x[b,s,:] = E[tokens[b,s],:] + P[s,:]   (f32 residual stream)
// ---------------------------------------------------------------------------
__global__ __launch_bounds__(256) void embed_k(const int* __restrict__ tok,
    const float* __restrict__ E, const float* __restrict__ P,
    float* __restrict__ x)
{
    const int bs = blockIdx.x;
    const int s  = bs & 511;
    const int t  = tok[bs];
    const int tid = threadIdx.x;
    const float* e = E + (size_t)t  * 768;
    const float* p = P + (size_t)s  * 768;
    float*       o = x + (size_t)bs * 768;
    o[tid]       = e[tid]       + p[tid];
    o[tid + 256] = e[tid + 256] + p[tid + 256];
    o[tid + 512] = e[tid + 512] + p[tid + 512];
}

// ---------------------------------------------------------------------------
// LayerNorm (768, biased var, eps=1e-5): f32 in, f16 out (GEMM A operand)
// ---------------------------------------------------------------------------
__global__ __launch_bounds__(256) void ln_k(const float* __restrict__ in,
    const float* __restrict__ g, const float* __restrict__ b,
    _Float16* __restrict__ out)
{
    __shared__ float rs[8], rq[8];
    const int row = blockIdx.x, tid = threadIdx.x;
    const float* p = in + (size_t)row * 768;
    float v0 = p[tid], v1 = p[tid + 256], v2 = p[tid + 512];
    float s  = v0 + v1 + v2;
    float sq = v0 * v0 + v1 * v1 + v2 * v2;
    #pragma unroll
    for (int off = 16; off > 0; off >>= 1) {
        s  += __shfl_xor(s,  off, 32);
        sq += __shfl_xor(sq, off, 32);
    }
    if ((tid & 31) == 0) { rs[tid >> 5] = s; rq[tid >> 5] = sq; }
    __syncthreads();
    float ts = 0.f, tq = 0.f;
    #pragma unroll
    for (int i = 0; i < 8; ++i) { ts += rs[i]; tq += rq[i]; }
    const float mean = ts * (1.0f / 768.0f);
    const float var  = tq * (1.0f / 768.0f) - mean * mean;
    const float inv  = rsqrtf(var + 1e-5f);
    _Float16* q = out + (size_t)row * 768;
    q[tid]       = (_Float16)(g[tid]       * (v0 - mean) * inv + b[tid]);
    q[tid + 256] = (_Float16)(g[tid + 256] * (v1 - mean) * inv + b[tid + 256]);
    q[tid + 512] = (_Float16)(g[tid + 512] * (v2 - mean) * inv + b[tid + 512]);
}

// ---------------------------------------------------------------------------
// In-place masked softmax over rows of [B*H*S, S] (f16 storage, f32 math):
//   val = (raw + (j >= i ? -1e10 : 0)) * (1/sqrt(64))  (mask incl. diagonal)
// ---------------------------------------------------------------------------
__global__ __launch_bounds__(256) void softmax_k(_Float16* __restrict__ sc)
{
    __shared__ float red[8];
    const int row = blockIdx.x;
    const int i   = row & 511;
    const int tid = threadIdx.x;
    _Float16* p = sc + (size_t)row * 512;
    const int j0 = tid, j1 = tid + 256;
    float a0 = ((float)p[j0] + (j0 >= i ? -1e10f : 0.f)) * 0.125f;
    float a1 = ((float)p[j1] + (j1 >= i ? -1e10f : 0.f)) * 0.125f;
    float mx = fmaxf(a0, a1);
    #pragma unroll
    for (int off = 16; off > 0; off >>= 1) mx = fmaxf(mx, __shfl_xor(mx, off, 32));
    if ((tid & 31) == 0) red[tid >> 5] = mx;
    __syncthreads();
    float gm = red[0];
    #pragma unroll
    for (int k = 1; k < 8; ++k) gm = fmaxf(gm, red[k]);
    float e0 = __expf(a0 - gm), e1 = __expf(a1 - gm);
    float s = e0 + e1;
    #pragma unroll
    for (int off = 16; off > 0; off >>= 1) s += __shfl_xor(s, off, 32);
    __syncthreads();
    if ((tid & 31) == 0) red[tid >> 5] = s;
    __syncthreads();
    float ts = 0.f;
    #pragma unroll
    for (int k = 0; k < 8; ++k) ts += red[k];
    const float r = 1.0f / ts;
    p[j0] = (_Float16)(e0 * r);
    p[j1] = (_Float16)(e1 * r);
}

// ---------------------------------------------------------------------------
extern "C" void kernel_launch(void* const* d_in, const int* in_sizes, int n_in,
                              void* d_out, int out_size, void* d_ws, size_t ws_size,
                              hipStream_t stream)
{
    const int*   tokens = (const int*)  d_in[0];
    const float* E   = (const float*)d_in[1];
    const float* P   = (const float*)d_in[2];
    const float* Wq  = (const float*)d_in[3];
    const float* Wk  = (const float*)d_in[4];
    const float* Wv  = (const float*)d_in[5];
    const float* Wo  = (const float*)d_in[6];
    const float* bo  = (const float*)d_in[7];
    const float* g1  = (const float*)d_in[8];
    const float* b1  = (const float*)d_in[9];
    const float* g2  = (const float*)d_in[10];
    const float* b2  = (const float*)d_in[11];
    const float* W1  = (const float*)d_in[12];
    const float* bf1 = (const float*)d_in[13];
    const float* W2  = (const float*)d_in[14];
    const float* bf2 = (const float*)d_in[15];
    const float* gf  = (const float*)d_in[16];
    const float* bfb = (const float*)d_in[17];
    float* out = (float*)d_out;

    const int Bt = 4, S = 512, D = 768, H = 12, HD = 64, DFF = 3072,
              V = 50257, L = 12;
    const size_t BSD  = (size_t)Bt * S * D;
    const size_t WSZ  = (size_t)L * D * D;       // per-proj weights, all layers
    const size_t FSZ  = (size_t)L * D * DFF;     // per-FF weights, all layers
    const size_t ESZ  = (size_t)V * D;

    float* x = (float*)d_ws;                     // [B,S,D] residual stream, f32
    _Float16* f16b = (_Float16*)(x + BSD);
    _Float16* h    = f16b;                 f16b += BSD;              // LN out
    _Float16* q    = f16b;                 f16b += BSD;
    _Float16* kx   = f16b;                 f16b += BSD;
    _Float16* vt   = f16b;                 f16b += BSD;              // [B,H,HD,S]
    _Float16* o    = f16b;                 f16b += BSD;
    _Float16* ff   = f16b;                 f16b += (size_t)Bt * S * DFF;
    _Float16* sc   = f16b;                 f16b += (size_t)Bt * H * S * S;
    _Float16* wq16 = f16b;                 f16b += WSZ;
    _Float16* wk16 = f16b;                 f16b += WSZ;
    _Float16* wv16 = f16b;                 f16b += WSZ;
    _Float16* wo16 = f16b;                 f16b += WSZ;
    _Float16* w116 = f16b;                 f16b += FSZ;
    _Float16* w216 = f16b;                 f16b += FSZ;
    _Float16* e16  = f16b;

    auto cvt = [&](const float* s, _Float16* d, long long n) {
        long long n4 = n >> 2;
        int blocks = (int)((n4 + 255) / 256); if (blocks > 4096) blocks = 4096;
        cvt_k<<<blocks, 256, 0, stream>>>(s, d, n4);
    };
    cvt(Wq, wq16, (long long)WSZ);
    cvt(Wk, wk16, (long long)WSZ);
    cvt(Wv, wv16, (long long)WSZ);
    cvt(Wo, wo16, (long long)WSZ);
    cvt(W1, w116, (long long)FSZ);
    cvt(W2, w216, (long long)FSZ);
    cvt(E,  e16,  (long long)ESZ);

    auto gemm = [&](const _Float16* A, const _Float16* Bm, const float* bias,
                    const float* R, void* C,
                    int M, int N, int K, int lda, int ldb, int ldc,
                    long long sa0, long long sa1, long long sb0, long long sb1,
                    long long sc0, long long sc1, int nb, int nh,
                    int op, int cfmt, int ct) {
        dim3 grid((N + 127) / 128, (M + 127) / 128, nb);
        hipLaunchKernelGGL(gemm_h, grid, dim3(256), 0, stream,
                           A, Bm, bias, R, C, M, N, K, lda, ldb, ldc,
                           sa0, sa1, sb0, sb1, sc0, sc1, nh, op, cfmt, ct);
    };

    embed_k<<<Bt * S, 256, 0, stream>>>(tokens, E, P, x);

    for (int l = 0; l < L; ++l) {
        const _Float16* wq = wq16 + (size_t)l * D * D;
        const _Float16* wk = wk16 + (size_t)l * D * D;
        const _Float16* wv = wv16 + (size_t)l * D * D;
        const _Float16* wo = wo16 + (size_t)l * D * D;
        const _Float16* w1 = w116 + (size_t)l * DFF * D;
        const _Float16* w2 = w216 + (size_t)l * D * DFF;

        // LN1 -> h (f16)
        ln_k<<<Bt * S, 256, 0, stream>>>(x, g1 + l * D, b1 + l * D, h);
        // Q, K projections (f16 out)
        gemm(h, wq, nullptr, nullptr, q,  Bt * S, D, D, D, D, D,
             0, 0, 0, 0, 0, 0, 1, 1, 0, 1, 0);
        gemm(h, wk, nullptr, nullptr, kx, Bt * S, D, D, D, D, D,
             0, 0, 0, 0, 0, 0, 1, 1, 0, 1, 0);
        // V projection stored transposed per head: vt[b, h, k, s] (f16)
        gemm(h, wv, nullptr, nullptr, vt, S, D, D, D, D, S,
             (long long)S * D, 0, 0, 0, (long long)D * S, 0, Bt, 1, 0, 1, 1);
        // raw scores (pre-mask, O(1) values -> safe in f16), batched over B*H
        gemm(q, kx, nullptr, nullptr, sc, S, S, HD, D, D, S,
             (long long)S * D, HD, (long long)S * D, HD,
             (long long)H * S * S, (long long)S * S, Bt * H, H, 0, 1, 0);
        // masked softmax (in place, f16)
        softmax_k<<<Bt * H * S, 256, 0, stream>>>(sc);
        // o[b,q,h*HD+k] = P . V (f16 out)
        gemm(sc, vt, nullptr, nullptr, o, S, HD, S, S, S, D,
             (long long)H * S * S, (long long)S * S,
             (long long)D * S, (long long)HD * S,
             (long long)S * D, HD, Bt * H, H, 0, 1, 0);
        // out-proj + bias + residual -> x (f32, in place)
        gemm(o, wo, bo + l * D, x, x, Bt * S, D, D, D, D, D,
             0, 0, 0, 0, 0, 0, 1, 1, 1 | 4, 0, 0);
        // LN2 -> h (f16)
        ln_k<<<Bt * S, 256, 0, stream>>>(x, g2 + l * D, b2 + l * D, h);
        // FF1 + bias + exact GELU (f16 out)
        gemm(h, w1, bf1 + l * DFF, nullptr, ff, Bt * S, DFF, D, D, D, DFF,
             0, 0, 0, 0, 0, 0, 1, 1, 1 | 2, 1, 0);
        // FF2 + bias + residual -> x (f32)
        gemm(ff, w2, bf2 + l * D, x, x, Bt * S, D, DFF, DFF, DFF, D,
             0, 0, 0, 0, 0, 0, 1, 1, 1 | 4, 0, 0);
    }

    // final LN + weight-tied LM head (f32 logits; ragged N zero-filled by TDM
    // OOB handling / clamped loads, stores guarded)
    ln_k<<<Bt * S, 256, 0, stream>>>(x, gf, bfb, h);
    gemm(h, e16, nullptr, nullptr, out, Bt * S, V, D, D, D, V,
         0, 0, 0, 0, 0, 0, 1, 1, 0, 0, 0);
}
